// CausalSelfAttention_43800076484932
// MI455X (gfx1250) — matrix operand order repaired and checked by reading, not compiled, not run on hardware
//
#include <hip/hip_runtime.h>
#include <hip/hip_bf16.h>
#include <math.h>

// ---- CDNA5 (gfx1250) wave32 WMMA types ----
typedef __attribute__((ext_vector_type(16))) __bf16        v16bf;
typedef __attribute__((ext_vector_type(8)))  float         v8f;
typedef __attribute__((ext_vector_type(4)))  unsigned int  u32x4;
typedef __attribute__((ext_vector_type(4)))  int           i32x4;
typedef __attribute__((ext_vector_type(8)))  int           i32x8;

union FragAB { u32x4 q[2]; v16bf v; };

#define WMMA_BF16(a, b, c) \
  __builtin_amdgcn_wmma_f32_16x16x32_bf16(false, (a), false, (b), (short)0, (c), false, false)

#define Bsz 4
#define Sn  2048
#define Dn  1024
#define Hn  16
#define HDn 64

__device__ __forceinline__ unsigned short f2bf(float f) {
  unsigned int u = __builtin_bit_cast(unsigned int, f);
  unsigned int r = u + 0x7FFFu + ((u >> 16) & 1u);   // round-to-nearest-even
  return (unsigned short)(r >> 16);
}

// ---- DPP butterfly reductions across a 16-lane row (no LDS, pure VALU) ----
// steps: quad_perm(1,0,3,2)=0xB1, quad_perm(2,3,0,1)=0x4E,
//        row_half_mirror=0x141, row_mirror=0x140
template <int CTRL>
__device__ __forceinline__ float dpp_xchg(float x) {
  int p = __builtin_amdgcn_update_dpp(0, __builtin_bit_cast(int, x), CTRL, 0xF, 0xF, true);
  return __builtin_bit_cast(float, p);
}
__device__ __forceinline__ float row_max16(float x) {
  x = fmaxf(x, dpp_xchg<0xB1>(x));
  x = fmaxf(x, dpp_xchg<0x4E>(x));
  x = fmaxf(x, dpp_xchg<0x141>(x));
  x = fmaxf(x, dpp_xchg<0x140>(x));
  return x;
}
__device__ __forceinline__ float row_sum16(float x) {
  x += dpp_xchg<0xB1>(x);
  x += dpp_xchg<0x4E>(x);
  x += dpp_xchg<0x141>(x);
  x += dpp_xchg<0x140>(x);
  return x;
}

// ---- Tensor Data Mover: 2D tile (tile_w x tile_h elements, 2B each) -> LDS ----
__device__ __forceinline__ void tdm_load_2d(const void* gsrc, unsigned lds_off,
                                            unsigned tensor_w, unsigned tensor_h,
                                            unsigned tile_w, unsigned tile_h,
                                            unsigned stride_w) {
  unsigned long long ga = (unsigned long long)gsrc;
  u32x4 g0 = { 1u,                                    // count=1, user mode
               lds_off,                               // LDS byte address
               (unsigned)ga,                          // global_addr[31:0]
               ((unsigned)(ga >> 32) & 0x01FFFFFFu) | 0x80000000u };  // [56:32] | type=2
  i32x8 g1 = { (int)(1u << 16),                                        // data_size=2B
               (int)((tensor_w & 0xFFFFu) << 16),                      // dim0[15:0]
               (int)(((tensor_w >> 16) & 0xFFFFu) | ((tensor_h & 0xFFFFu) << 16)),
               (int)(((tensor_h >> 16) & 0xFFFFu) | (tile_w << 16)),   // dim1 hi | tile_dim0
               (int)(tile_h & 0xFFFFu),                                // tile_dim1
               (int)stride_w, 0, 0 };                                  // dim0 stride
  i32x4 z4 = { 0, 0, 0, 0 };
#if defined(__clang_major__) && (__clang_major__ >= 23)
  i32x8 z8 = { 0, 0, 0, 0, 0, 0, 0, 0 };
  __builtin_amdgcn_tensor_load_to_lds(g0, g1, z4, z4, z8, 0);
#else
  __builtin_amdgcn_tensor_load_to_lds(g0, g1, z4, z4, 0);
#endif
}

// ---------------- conversion kernels ----------------
__global__ void k_cvt(const float* __restrict__ src, unsigned short* __restrict__ dst, int n) {
  int i = blockIdx.x * blockDim.x + threadIdx.x;
  if (i < n) dst[i] = f2bf(src[i]);
}

// W[K][N] (fp32) -> Wt[N][K] (bf16)
__global__ void k_cvt_t(const float* __restrict__ W, unsigned short* __restrict__ Wt, int K, int N) {
  long long i = (long long)blockIdx.x * blockDim.x + threadIdx.x;
  long long total = (long long)K * N;
  if (i < total) {
    int n = (int)(i / K);
    int k = (int)(i % K);
    Wt[i] = f2bf(W[(long long)k * N + n]);
  }
}

// ---------------- tiled WMMA GEMM, TDM double-buffered ----------------
template <int MODE>
__global__ __launch_bounds__(256)
void k_gemm(const unsigned short* __restrict__ A, const unsigned short* __restrict__ Bt,
            const float* __restrict__ bias, float* __restrict__ Cf,
            unsigned short* __restrict__ Ck, unsigned short* __restrict__ Cq,
            unsigned short* __restrict__ Cv, int Kd) {
  __shared__ __align__(16) unsigned short As[2][128 * 32];  // 2 x 8 KB
  __shared__ __align__(16) unsigned short Bs[2][64 * 32];   // 2 x 4 KB
  const int t = threadIdx.x;
  const int wid = t >> 5, lane = t & 31;
  const int m0 = blockIdx.y * 128, n0 = blockIdx.x * 64;
  const int mm = lane & 15, ci = lane >> 4;

  const unsigned asOff[2] = { (unsigned)(unsigned long long)(uintptr_t)&As[0][0],
                              (unsigned)(unsigned long long)(uintptr_t)&As[1][0] };
  const unsigned bsOff[2] = { (unsigned)(unsigned long long)(uintptr_t)&Bs[0][0],
                              (unsigned)(unsigned long long)(uintptr_t)&Bs[1][0] };

  v8f acc[4] = {};
  const int nsteps = Kd >> 5;

  if (wid == 0) {  // uniform per wave: waves 1..7 branch over, never issue
    tdm_load_2d(A + (size_t)m0 * Kd, asOff[0], (unsigned)Kd, 1u << 20, 32, 128, (unsigned)Kd);
    tdm_load_2d(Bt + (size_t)n0 * Kd, bsOff[0], (unsigned)Kd, 1u << 20, 32, 64, (unsigned)Kd);
  }

  for (int s = 0; s < nsteps; ++s) {
    const int buf = s & 1;
    if (wid == 0) {
      if (s + 1 < nsteps) {
        const int kk = (s + 1) << 5;
        tdm_load_2d(A + (size_t)m0 * Kd + kk, asOff[buf ^ 1], (unsigned)Kd, 1u << 20, 32, 128, (unsigned)Kd);
        tdm_load_2d(Bt + (size_t)n0 * Kd + kk, bsOff[buf ^ 1], (unsigned)Kd, 1u << 20, 32, 64, (unsigned)Kd);
        __builtin_amdgcn_s_wait_tensorcnt((short)2);  // stage s done; s+1 in flight
      } else {
        __builtin_amdgcn_s_wait_tensorcnt((short)0);
      }
    }
    __syncthreads();

    // A fragment 16x32: lanes 0-15 K=0..15, lanes 16-31 K=16..31
    FragAB a;
    const u32x4* ap = (const u32x4*)(&As[buf][0] + (wid * 16 + mm) * 32);
    a.q[0] = ap[2 * ci];
    a.q[1] = ap[2 * ci + 1];

#pragma unroll
    for (int nt = 0; nt < 4; ++nt) {
      // B fragment 32x16: lane column N, lanes 0-15 K=0..15, lanes 16-31 K=16..31
      FragAB b;
      const u32x4* bp = (const u32x4*)(&Bs[buf][0] + (nt * 16 + mm) * 32 + ci * 16);
      b.q[0] = bp[0];
      b.q[1] = bp[1];
      acc[nt] = WMMA_BF16(a.v, b.v, acc[nt]);
    }
    __syncthreads();
  }

  // C layout: VGPR i -> row (i + 8*(lane/16)), col = lane%16
  const int mrow = m0 + wid * 16 + ci * 8;
#pragma unroll
  for (int nt = 0; nt < 4; ++nt) {
#pragma unroll
    for (int i = 0; i < 8; ++i) {
      int m = mrow + i;
      int n = n0 + nt * 16 + mm;
      float v = acc[nt][i] + bias[n];
      if (MODE == 0) {
        int sect = n >> 10, col = n & 1023;       // chunk order: 0=k, 1=q, 2=v
        int h = col >> 6, hd = col & 63;
        int b = m >> 11, s = m & 2047;
        unsigned short bv = f2bf(v);
        size_t bhv = (size_t)b * Hn + h;
        if (sect == 0)      Ck[(bhv * Sn + s) * HDn + hd] = bv;
        else if (sect == 1) Cq[(bhv * Sn + s) * HDn + hd] = bv;
        else                Cv[(bhv * HDn + hd) * Sn + s] = bv;  // V transposed for PV B-frags
      } else {
        Cf[(size_t)m * Dn + n] = v;
      }
    }
  }
}

// ---------------- flash-attention with WMMA ----------------
__global__ __launch_bounds__(256)
void k_attn(const unsigned short* __restrict__ Qb, const unsigned short* __restrict__ Kb,
            const unsigned short* __restrict__ Vt, unsigned short* __restrict__ AO) {
  __shared__ __align__(16) unsigned short Ps[8][16 * 32];  // 1 KB per wave
  const int t = threadIdx.x, wid = t >> 5, lane = t & 31;
  const int bh = blockIdx.x;
  const int qbase = blockIdx.y * 128 + wid * 16;
  const unsigned short* Qp = Qb + (size_t)bh * Sn * HDn;
  const unsigned short* Kp = Kb + (size_t)bh * Sn * HDn;
  const unsigned short* Vp = Vt + (size_t)bh * HDn * Sn;
  const int mm = lane & 15, hi = lane >> 4, ci = hi;

  // Q fragments for the two K-steps over HD=64
  FragAB aq0, aq1;
  {
    const u32x4* qr = (const u32x4*)(Qp + (size_t)(qbase + mm) * HDn);
    aq0.q[0] = qr[2 * ci];     aq0.q[1] = qr[2 * ci + 1];
    aq1.q[0] = qr[4 + 2 * ci]; aq1.q[1] = qr[5 + 2 * ci];
  }

  float rm[8], rl[8];
#pragma unroll
  for (int i = 0; i < 8; ++i) { rm[i] = -__builtin_inff(); rl[i] = 0.f; }
  v8f oacc[4] = {};
  const int qmax = qbase + 15;

  auto kblock = [&](int kb0, bool MASK) {
    v8f sc0 = {}, sc1 = {};
    {  // score tile 0: keys kb0..kb0+15
      const u32x4* kp = (const u32x4*)(Kp + (size_t)(kb0 + mm) * HDn + ci * 16);
      FragAB b; b.q[0] = kp[0]; b.q[1] = kp[1];
      sc0 = WMMA_BF16(aq0.v, b.v, sc0);
      const u32x4* kp2 = (const u32x4*)(Kp + (size_t)(kb0 + mm) * HDn + 32 + ci * 16);
      FragAB b2; b2.q[0] = kp2[0]; b2.q[1] = kp2[1];
      sc0 = WMMA_BF16(aq1.v, b2.v, sc0);
    }
    {  // score tile 1: keys kb0+16..kb0+31
      const u32x4* kp = (const u32x4*)(Kp + (size_t)(kb0 + 16 + mm) * HDn + ci * 16);
      FragAB b; b.q[0] = kp[0]; b.q[1] = kp[1];
      sc1 = WMMA_BF16(aq0.v, b.v, sc1);
      const u32x4* kp2 = (const u32x4*)(Kp + (size_t)(kb0 + 16 + mm) * HDn + 32 + ci * 16);
      FragAB b2; b2.q[0] = kp2[0]; b2.q[1] = kp2[1];
      sc1 = WMMA_BF16(aq1.v, b2.v, sc1);
    }

    const float scale = 0.125f;  // 1/sqrt(64)
    float al8[8], p0a[8], p1a[8];
#pragma unroll
    for (int i = 0; i < 8; ++i) {
      float s0 = sc0[i] * scale, s1 = sc1[i] * scale;
      if (MASK) {
        const int q = qbase + i + 8 * hi;
        if (kb0 + mm > q)      s0 = -__builtin_inff();   // causal mask
        if (kb0 + 16 + mm > q) s1 = -__builtin_inff();
      }
      float mx = row_max16(fmaxf(s0, s1));          // DPP butterfly, no LDS
      float mn = fmaxf(rm[i], mx);
      float al = __expf(rm[i] - mn);
      float p0 = __expf(s0 - mn), p1 = __expf(s1 - mn);
      float ps = row_sum16(p0 + p1);                // DPP butterfly, no LDS
      rl[i] = rl[i] * al + ps;
      rm[i] = mn;
      al8[i] = al; p0a[i] = p0; p1a[i] = p1;
    }
#pragma unroll
    for (int nt = 0; nt < 4; ++nt)
#pragma unroll
      for (int i = 0; i < 8; ++i) oacc[nt][i] *= al8[i];

    // restripe P (C-layout f32 -> A-layout bf16) through per-wave LDS scratch
    unsigned short* ps = Ps[wid];
#pragma unroll
    for (int i = 0; i < 8; ++i) {
      int row = i + 8 * hi;
      ps[row * 32 + mm]      = f2bf(p0a[i]);
      ps[row * 32 + 16 + mm] = f2bf(p1a[i]);
    }
    asm volatile("s_wait_dscnt 0" ::: "memory");  // per-wave LDS RAW; waves diverge, no barrier
    FragAB pa;
    {
      const u32x4* pp = (const u32x4*)(ps + mm * 32);
      pa.q[0] = pp[2 * ci];
      pa.q[1] = pp[2 * ci + 1];
    }
    // PV: B-frags contiguous from transposed V
#pragma unroll
    for (int nt = 0; nt < 4; ++nt) {
      const u32x4* vp = (const u32x4*)(Vp + (size_t)(nt * 16 + mm) * Sn + kb0 + ci * 16);
      FragAB bv; bv.q[0] = vp[0]; bv.q[1] = vp[1];
      oacc[nt] = WMMA_BF16(pa.v, bv.v, oacc[nt]);
    }
    __builtin_prefetch(Kp + (size_t)(kb0 + 32) * HDn, 0, 1);  // global_prefetch next block
  };

  int kb0 = 0;
  for (; kb0 + 31 <= qbase; kb0 += 32) kblock(kb0, false);  // full blocks: no mask VALU
  for (; kb0 <= qmax; kb0 += 32) kblock(kb0, true);         // diagonal block(s)

  // finalize and store attn output as bf16 [b, s, h, hd] (= reshape layout for out-proj A)
  const int b = bh >> 4, h = bh & 15;
#pragma unroll
  for (int nt = 0; nt < 4; ++nt) {
#pragma unroll
    for (int i = 0; i < 8; ++i) {
      int q = qbase + i + 8 * hi;
      int hd = nt * 16 + mm;
      float v = oacc[nt][i] / rl[i];
      AO[(((size_t)b * Sn + q) * Hn + h) * HDn + hd] = f2bf(v);
    }
  }
}

// ---------------- host launch ----------------
extern "C" void kernel_launch(void* const* d_in, const int* in_sizes, int n_in,
                              void* d_out, int out_size, void* d_ws, size_t ws_size,
                              hipStream_t stream) {
  (void)in_sizes; (void)n_in; (void)out_size; (void)ws_size;
  const float* x     = (const float*)d_in[0];
  const float* W_in  = (const float*)d_in[1];
  const float* b_in  = (const float*)d_in[2];
  const float* W_out = (const float*)d_in[3];
  const float* b_out = (const float*)d_in[4];
  float* out = (float*)d_out;

  char* ws = (char*)d_ws;
  size_t off = 0;
  auto carve = [&](size_t bytes) {
    void* p = ws + off;
    off += (bytes + 255) & ~(size_t)255;
    return p;
  };
  const size_t nTok = (size_t)Bsz * Sn;                // 8192
  unsigned short* xb    = (unsigned short*)carve(nTok * Dn * 2);
  unsigned short* WinT  = (unsigned short*)carve((size_t)3 * Dn * Dn * 2);
  unsigned short* WoutT = (unsigned short*)carve((size_t)Dn * Dn * 2);
  unsigned short* Kb    = (unsigned short*)carve(nTok * Dn * 2);
  unsigned short* Qb    = (unsigned short*)carve(nTok * Dn * 2);
  unsigned short* Vt    = (unsigned short*)carve(nTok * Dn * 2);
  unsigned short* AO    = (unsigned short*)carve(nTok * Dn * 2);

  // 1) fp32 -> bf16 conversions (weights pre-transposed to [N][K])
  k_cvt<<<(int)(nTok * Dn / 256), 256, 0, stream>>>(x, xb, (int)(nTok * Dn));
  k_cvt_t<<<(3 * Dn * Dn) / 256, 256, 0, stream>>>(W_in, WinT, Dn, 3 * Dn);
  k_cvt_t<<<(Dn * Dn) / 256, 256, 0, stream>>>(W_out, WoutT, Dn, Dn);

  // 2) QKV projection: [8192,1024] x [1024,3072] -> scatter K/Q/Vt bf16
  k_gemm<0><<<dim3(3 * Dn / 64, (int)(nTok / 128)), 256, 0, stream>>>(
      xb, WinT, b_in, nullptr, Kb, Qb, Vt, Dn);

  // 3) causal flash attention per (b,h), 128 query rows per block
  k_attn<<<dim3(Bsz * Hn, Sn / 128), 256, 0, stream>>>(Qb, Kb, Vt, AO);

  // 4) output projection: [8192,1024] x [1024,1024] + bias -> fp32 out
  k_gemm<1><<<dim3(Dn / 64, (int)(nTok / 128)), 256, 0, stream>>>(
      AO, WoutT, b_out, out, nullptr, nullptr, nullptr, Dn);
}
